// MutualAttentionFusion_17772574671579
// MI455X (gfx1250) — matrix operand order repaired and checked by reading, not compiled, not run on hardware
//
#include <hip/hip_runtime.h>

#define B_  8
#define C_  256
#define C8_ 32
#define L_  2048

typedef __attribute__((ext_vector_type(16))) _Float16 v16h;
typedef __attribute__((ext_vector_type(8)))  float    v8f;
typedef __attribute__((ext_vector_type(4)))  float    v4f;

union V16H { v16h h; v4f f[2]; };

// v_max_num_f32 quiets NaNs itself; bypass clang's redundant canonicalize ops.
__device__ __forceinline__ float vmax(float a, float b) {
    float d;
    asm("v_max_num_f32 %0, %1, %2" : "=v"(d) : "v"(a), "v"(b));
    return d;
}

// Butterfly reduction across the 16 lanes of each half-wave row group.
// DPP16 as a VOP2 source modifier: one instruction per step, no canonicalize,
// no LDS. quad_perm[1,0,3,2]=xor1, quad_perm[2,3,0,1]=xor2, row_half_mirror
// and row_mirror complete the tree once quads/octets are uniform.
__device__ __forceinline__ float row16_max(float v) {
    float r;
    asm("v_max_num_f32_dpp %0, %1, %1 quad_perm:[1,0,3,2] row_mask:0xf bank_mask:0xf bound_ctrl:1"
        : "=&v"(r) : "v"(v));
    asm("v_max_num_f32_dpp %0, %1, %1 quad_perm:[2,3,0,1] row_mask:0xf bank_mask:0xf bound_ctrl:1"
        : "=&v"(v) : "v"(r));
    asm("v_max_num_f32_dpp %0, %1, %1 row_half_mirror row_mask:0xf bank_mask:0xf bound_ctrl:1"
        : "=&v"(r) : "v"(v));
    asm("v_max_num_f32_dpp %0, %1, %1 row_mirror row_mask:0xf bank_mask:0xf bound_ctrl:1"
        : "=&v"(v) : "v"(r));
    return v;
}
__device__ __forceinline__ float row16_sum(float v) {
    float r;
    asm("v_add_f32_dpp %0, %1, %1 quad_perm:[1,0,3,2] row_mask:0xf bank_mask:0xf bound_ctrl:1"
        : "=&v"(r) : "v"(v));
    asm("v_add_f32_dpp %0, %1, %1 quad_perm:[2,3,0,1] row_mask:0xf bank_mask:0xf bound_ctrl:1"
        : "=&v"(v) : "v"(r));
    asm("v_add_f32_dpp %0, %1, %1 row_half_mirror row_mask:0xf bank_mask:0xf bound_ctrl:1"
        : "=&v"(r) : "v"(v));
    asm("v_add_f32_dpp %0, %1, %1 row_mirror row_mask:0xf bank_mask:0xf bound_ctrl:1"
        : "=&v"(v) : "v"(r));
    return v;
}

// ---------------------------------------------------------------------------
// Projection: Q = Wq x + bq, K = Wk y + bk  (stored transposed [b][l][c8], f16)
// ---------------------------------------------------------------------------
__global__ __launch_bounds__(256)
void proj_qk(const float* __restrict__ x, const float* __restrict__ y,
             const float* __restrict__ Wq, const float* __restrict__ bq,
             const float* __restrict__ Wk, const float* __restrict__ bk,
             _Float16* __restrict__ Qt, _Float16* __restrict__ Kt)
{
    int tid   = blockIdx.x * blockDim.x + threadIdx.x;   // [which][b][o][l]
    int lpos  = tid & (L_ - 1);
    int o     = (tid >> 11) & (C8_ - 1);
    int b     = (tid >> 16) & (B_ - 1);
    int which = tid >> 19;

    const float* in   = which ? y  : x;
    const float* W    = which ? Wk : Wq;
    const float* bias = which ? bk : bq;

    float acc = bias[o];
    const float* xr = in + b * (C_ * L_) + lpos;
    const float* wr = W + o * C_;
#pragma unroll 8
    for (int c = 0; c < C_; ++c)
        acc += wr[c] * xr[c * L_];

    _Float16* T = which ? Kt : Qt;
    T[(b * L_ + lpos) * C8_ + o] = (_Float16)acc;
}

// ---------------------------------------------------------------------------
// Projection: V = Wv y + bv  (natural layout [b][c][l], f16)
// ---------------------------------------------------------------------------
__global__ __launch_bounds__(256)
void proj_v(const float* __restrict__ y,
            const float* __restrict__ Wv, const float* __restrict__ bv,
            _Float16* __restrict__ Vh)
{
    int tid  = blockIdx.x * blockDim.x + threadIdx.x;    // [b][c][l]
    int lpos = tid & (L_ - 1);
    int o    = (tid >> 11) & (C_ - 1);
    int b    = tid >> 19;

    float acc = bv[o];
    const float* yr = y + b * (C_ * L_) + lpos;
    const float* wr = Wv + o * C_;
#pragma unroll 8
    for (int c = 0; c < C_; ++c)
        acc += wr[c] * yr[c * L_];

    Vh[tid] = (_Float16)acc;   // tid == (b*C_ + o)*L_ + lpos
}

// ---------------------------------------------------------------------------
// Flash attention: one wave per (16 query rows x 128 channels).
//   S = Q Kt   (v_wmma_f32_16x16x32_f16, two 16-col subtiles per 32-j step)
//   online softmax in C/D lane layout; row reductions via fused DPP16 VOP2
//   O += P V   (8 channel-tile WMMAs, f32 accumulate)
//   out = x + gamma * O / l
// ---------------------------------------------------------------------------
__global__ __launch_bounds__(32)
void flash_attn(const _Float16* __restrict__ Qt,
                const _Float16* __restrict__ Kt,
                const _Float16* __restrict__ Vh,
                const float* __restrict__ x,
                const float* __restrict__ gamma,
                float* __restrict__ out)
{
    __shared__ _Float16 Pl[16 * 32];     // P tile staging (C/D layout -> A layout)

    const int lane = threadIdx.x;
    const int l16  = lane & 15;
    const int lh   = lane >> 4;          // 0: rows 0-7 / K 0-15 half, 1: rows 8-15 / K 16-31 half

    int t  = blockIdx.x;
    const int cs = t & 1;  t >>= 1;      // channel split: 0 -> c 0..127, 1 -> c 128..255
    const int b  = t >> 7;               // L/16 = 128 i-tiles per batch
    const int i0 = (t & 127) << 4;

    // ---- A operand: Q tile (16 x 32 f16), row-major rows of 32 channels ----
    V16H aq;
    {
        const int qoff = (b * L_ + i0 + l16) * C8_ + lh * 16;
        aq.f[0] = *(const v4f*)(Qt + qoff);        // halves 0-7
        aq.f[1] = *(const v4f*)(Qt + qoff + 8);    // halves 8-15
    }

    float m[8], l[8];
    v8f   acc[8];
    v8f   zero = {};
#pragma unroll
    for (int r = 0; r < 8; ++r) { m[r] = -3.0e38f; l[r] = 0.0f; }
#pragma unroll
    for (int ct = 0; ct < 8; ++ct) acc[ct] = zero;

    const float LOG2E = 1.4426950408889634f;

    for (int j0 = 0; j0 < L_; j0 += 32) {
        // ---- B operands: K^T (32 chan x 16 j), column j's channels contiguous ----
        V16H bk0, bk1;
        {
            const int koff = (b * L_ + j0 + l16) * C8_ + lh * 16;
            bk0.f[0] = *(const v4f*)(Kt + koff);
            bk0.f[1] = *(const v4f*)(Kt + koff + 8);
            bk1.f[0] = *(const v4f*)(Kt + koff + 16 * C8_);
            bk1.f[1] = *(const v4f*)(Kt + koff + 16 * C8_ + 8);
        }

        v8f s0 = __builtin_amdgcn_wmma_f32_16x16x32_f16(false, aq.h, false, bk0.h,
                                                        (short)0, zero, false, false);
        v8f s1 = __builtin_amdgcn_wmma_f32_16x16x32_f16(false, aq.h, false, bk1.h,
                                                        (short)0, zero, false, false);

        // ---- online softmax: each lane owns 8 rows (M = r + 8*lh), col N = l16 ----
        float p0[8], p1[8], corr[8];
#pragma unroll
        for (int r = 0; r < 8; ++r) {
            float rm = row16_max(vmax(s0[r], s1[r]));
            float mn = vmax(m[r], rm);
            corr[r] = __builtin_exp2f((m[r]  - mn) * LOG2E);
            p0[r]   = __builtin_exp2f((s0[r] - mn) * LOG2E);
            p1[r]   = __builtin_exp2f((s1[r] - mn) * LOG2E);
            float rs = row16_sum(p0[r] + p1[r]);
            l[r] = l[r] * corr[r] + rs;
            m[r] = mn;
        }
#pragma unroll
        for (int ct = 0; ct < 8; ++ct)
#pragma unroll
            for (int r = 0; r < 8; ++r) acc[ct][r] *= corr[r];

        // ---- P: C/D layout -> row-major LDS -> A layout ----
        __syncthreads();                       // WAR vs previous iteration's P loads
#pragma unroll
        for (int r = 0; r < 8; ++r) {
            int M = r + 8 * lh;
            Pl[M * 32 + l16]      = (_Float16)p0[r];
            Pl[M * 32 + 16 + l16] = (_Float16)p1[r];
        }
        __syncthreads();

        V16H ap;
        {
            const _Float16* prow = &Pl[l16 * 32 + lh * 16];
            ap.f[0] = *(const v4f*)(prow);
            ap.f[1] = *(const v4f*)(prow + 8);
        }

        // ---- O += P(16x32) * V(32j x 16c), per channel tile ----
#pragma unroll
        for (int ct = 0; ct < 8; ++ct) {
            const int cabs = cs * 128 + ct * 16 + l16;      // B column = channel
            const int voff = (b * C_ + cabs) * L_ + j0 + lh * 16;
            V16H bv;
            bv.f[0] = *(const v4f*)(Vh + voff);
            bv.f[1] = *(const v4f*)(Vh + voff + 8);
            acc[ct] = __builtin_amdgcn_wmma_f32_16x16x32_f16(false, ap.h, false, bv.h,
                                                             (short)0, acc[ct], false, false);
        }
    }

    // ---- epilogue: out = x + gamma * O / l ----
    const float g = gamma[0];
    float inv[8];
#pragma unroll
    for (int r = 0; r < 8; ++r) inv[r] = 1.0f / l[r];

#pragma unroll
    for (int ct = 0; ct < 8; ++ct) {
        const int cabs = cs * 128 + ct * 16 + l16;
        const int base = (b * C_ + cabs) * L_ + i0 + 8 * lh;  // rows contiguous in l
        v4f x0 = *(const v4f*)(x + base);
        v4f x1 = *(const v4f*)(x + base + 4);
        v4f o0, o1;
#pragma unroll
        for (int r = 0; r < 4; ++r) {
            o0[r] = x0[r] + g * acc[ct][r]     * inv[r];
            o1[r] = x1[r] + g * acc[ct][r + 4] * inv[r + 4];
        }
        *(v4f*)(out + base)     = o0;
        *(v4f*)(out + base + 4) = o1;
    }
}

// ---------------------------------------------------------------------------
extern "C" void kernel_launch(void* const* d_in, const int* in_sizes, int n_in,
                              void* d_out, int out_size, void* d_ws, size_t ws_size,
                              hipStream_t stream)
{
    const float* x     = (const float*)d_in[0];
    const float* y     = (const float*)d_in[1];
    const float* Wq    = (const float*)d_in[2];
    const float* bq    = (const float*)d_in[3];
    const float* Wk    = (const float*)d_in[4];
    const float* bk    = (const float*)d_in[5];
    const float* Wv    = (const float*)d_in[6];
    const float* bv    = (const float*)d_in[7];
    const float* gamma = (const float*)d_in[8];
    float* out = (float*)d_out;

    _Float16* Qt = (_Float16*)d_ws;                       // 1 MB
    _Float16* Kt = Qt + (size_t)B_ * L_ * C8_;            // 1 MB
    _Float16* Vh = Kt + (size_t)B_ * L_ * C8_;            // 8 MB

    proj_qk<<<(2 * B_ * C8_ * L_) / 256, 256, 0, stream>>>(x, y, Wq, bq, Wk, bk, Qt, Kt);
    proj_v <<<(B_ * C_ * L_) / 256, 256, 0, stream>>>(y, Wv, bv, Vh);
    flash_attn<<<B_ * (L_ / 16) * 2, 32, 0, stream>>>(Qt, Kt, Vh, x, gamma, out);
}
